// FM_38560216383906
// MI455X (gfx1250) — compile-verified, matrix operand order checked by reading, not run on hardware
//
#include <hip/hip_runtime.h>
#include <hip/hip_bf16.h>

#define COLS  22
#define VOCAB 12
#define DIM   64
#define BATCH 2048
#define KID   1408   // COLS*DIM
#define NCH   44     // KID/32
#define NTILE 128    // BATCH/16

typedef __attribute__((ext_vector_type(16))) __bf16 v16bf;
typedef __attribute__((ext_vector_type(8)))  float  v8f;
typedef __attribute__((ext_vector_type(4)))  int    v4i;
typedef __attribute__((address_space(1))) v4i* gv4i_p;  // global int4*
typedef __attribute__((address_space(3))) v4i* lv4i_p;  // LDS int4*

// workspace layout (bytes, all 32B-aligned)
#define OFF_T      0u          // float  [2048][1408]            11,534,336
#define OFF_APACK  11534336u   // bf16 A-operand tiles (pairsum)  5,767,168
#define OFF_AMULT  17301504u   // bf16 A-operand tiles (S_mult)   8,388,608
#define OFF_BPAIR  25690112u   // bf16 B-operand (combined Wc)       45,056
#define OFF_BMULT  25735168u   // bf16 B-operand (W_mult tiles)     262,144
#define OFF_TCD    25997312u   // f32 T in C/D lane layout       16,777,216
#define OFF_PART   42774528u   // float [128] per-tile loss partials

#if defined(__AMDGCN__) && __has_builtin(__builtin_amdgcn_global_load_async_to_lds_b128)
#define ASYNC_BMULT 1
#endif

// 16-bit A/B operand lane layout: lane<16 -> K in {0..7} (idx 0..7) and
// {16..23} (idx 8..15); lane>=16 -> K in {8..15} and {24..31}.
__device__ __forceinline__ int kpos_K(int lane, int idx) {
  int h = lane >> 4;
  return (idx < 8) ? (idx + 8 * h) : ((idx - 8) + 16 + 8 * h);
}
__device__ __forceinline__ void inv_kpos(int r, int L0, int& lane, int& idx) {
  if (r < 8)       { lane = L0;      idx = r;      }
  else if (r < 16) { lane = L0 + 16; idx = r - 8;  }
  else if (r < 24) { lane = L0;      idx = r - 8;  }
  else             { lane = L0 + 16; idx = r - 16; }
}

// ---------------------------------------------------------------------------
// Pack combined linear pair weights into B-operand layout.
// Column n: 0,1=plus(k) 2,3=cat(k) 4,5=max-pairsum(k) 6,7=min-pairsum(k)
// ---------------------------------------------------------------------------
__global__ void k_pack_pair(const float* __restrict__ Wp,
                            const float* __restrict__ Wx,
                            const float* __restrict__ Wn,
                            const float* __restrict__ Wc,
                            __bf16* __restrict__ Bpair) {
  int e = blockIdx.x * blockDim.x + threadIdx.x;
  if (e >= NCH * 512) return;
  int idx  = e & 15;
  int lane = (e >> 4) & 31;
  int c    = e >> 9;
  int K  = kpos_K(lane, idx);
  int id = c * 32 + K;
  int i  = id / DIM, d = id % DIM;
  int n  = lane & 15;
  float v = 0.f;
  if (n < 8) {
    int k = n & 1;
    int t = n >> 1;  // 0=plus 1=cat 2=max 3=min
    for (int j = 0; j < COLS; ++j) {
      int ij = ((i * COLS + j) * 2 + k);
      int ji = ((j * COLS + i) * 2 + k);
      if (t == 0)      v += Wp[ij * DIM + d] + Wp[ji * DIM + d];
      else if (t == 1) v += Wc[ij * 2 * DIM + d] + Wc[ji * 2 * DIM + DIM + d];
      else if (t == 2) v += Wx[ij * DIM + d] + Wx[ji * DIM + d];
      else             v += Wn[ij * DIM + d] + Wn[ji * DIM + d];
    }
  }
  Bpair[e] = (__bf16)v;
}

// ---------------------------------------------------------------------------
// Pack W_mult[i,j,k,d] into per-(d,k,itile) 32(K=j)x16(N=i) B-operand tiles.
// Tile order within a d: (k*2 + it), 4 contiguous tiles = 4KB per d.
// ---------------------------------------------------------------------------
__global__ void k_pack_mult(const float* __restrict__ Wm,
                            __bf16* __restrict__ Bmult) {
  int e = blockIdx.x * blockDim.x + threadIdx.x;
  if (e >= DIM * 2 * 2 * 512) return;
  int idx  = e & 15;
  int lane = (e >> 4) & 31;
  int g    = e >> 9;          // ((d*2 + k)*2 + it)
  int it = g & 1, k = (g >> 1) & 1, d = g >> 2;
  int j = kpos_K(lane, idx);
  int i = it * 16 + (lane & 15);
  float v = 0.f;
  if (i < COLS && j < COLS) v = Wm[((i * COLS + j) * 2 + k) * DIM + d];
  Bmult[e] = (__bf16)v;
}

// ---------------------------------------------------------------------------
// T = tanh-MLP(embedding gather); writes f32 copy + pre-swizzled bf16
// A-operand arrays + f32 copy in WMMA C/D lane layout (for the diag dot).
// ---------------------------------------------------------------------------
__global__ void k_T(const int* __restrict__ fidx, const float* __restrict__ emb,
                    const float* __restrict__ w1, const float* __restrict__ b1,
                    const float* __restrict__ w2, const float* __restrict__ b2s,
                    float* __restrict__ Tf32, __bf16* __restrict__ Apack,
                    __bf16* __restrict__ Amult, float* __restrict__ TcD) {
  int e = blockIdx.x * blockDim.x + threadIdx.x;
  if (e >= BATCH * KID) return;
  int b = e / KID, rem = e % KID;
  int i = rem / DIM, d = rem % DIM;
  int vidx = fidx[b * COLS + i];
  float ev = emb[(i * VOCAB + vidx) * DIM + d];
  float t = b2s[0];
#pragma unroll
  for (int h = 0; h < 8; ++h) t += w2[h] * tanhf(ev * w1[h] + b1[h]);
  Tf32[(size_t)b * KID + rem] = t;
  __bf16 tb = (__bf16)t;
  int tile = b >> 4, L0 = b & 15;
  int lane, idx;
  // pairsum A-operand: K = flattened (i,d), 44 chunks of 32 (exactly 1408)
  inv_kpos(rem & 31, L0, lane, idx);
  Apack[(((size_t)tile * NCH + (rem >> 5)) * 32 + lane) * 16 + idx] = tb;
  // S_mult A-operand: per-d tile, K = j (= i), padded to 32
  inv_kpos(i, L0, lane, idx);
  Amult[(((size_t)tile * DIM + d) * 32 + lane) * 16 + idx] = tb;
  if (i == 0) {  // zero-fill K padding 22..31 once per (b,d)
    for (int r2 = COLS; r2 < 32; ++r2) {
      inv_kpos(r2, L0, lane, idx);
      Amult[(((size_t)tile * DIM + d) * 32 + lane) * 16 + idx] = (__bf16)0.f;
    }
  }
  // C/D-layout f32: M = b%16 -> (r = M%8, half = M/8), N = i%16, it = i/16
  {
    int m = b & 15, r = m & 7, hh = m >> 3;
    int it = i >> 4, nn = i & 15;
    int ln = nn + 16 * hh;
    TcD[((((size_t)tile * DIM + d) * 2 + it) * 32 + ln) * 8 + r] = t;
    if (i == 0) {  // zero-fill it=1 columns i=22..31 for this (r,hh) slice
      for (int nz = 6; nz < 16; ++nz) {
        int lz = nz + 16 * hh;
        TcD[((((size_t)tile * DIM + d) * 2 + 1) * 32 + lz) * 8 + r] = 0.f;
      }
    }
  }
}

// ---------------------------------------------------------------------------
// Main: 4 waves/block, one 16-batch WMMA tile per wave.
// ---------------------------------------------------------------------------
__global__ void k_main(const float* __restrict__ Tf32,
                       const __bf16* __restrict__ Apack,
                       const __bf16* __restrict__ Bpair,
                       const __bf16* __restrict__ Amult,
                       const __bf16* __restrict__ Bmult,
                       const float* __restrict__ TcD,
                       const float* __restrict__ Wmax,
                       const float* __restrict__ Wmin,
                       const float* __restrict__ label,
                       const float* __restrict__ posw,
                       const float* __restrict__ mixw,
                       float* __restrict__ out,
                       float* __restrict__ partials) {
  __shared__ float Dst[4][16][16];  // pairsum GEMM result staging
  __shared__ float qm[4][2][16];    // S_mult per-(k,b)
  __shared__ float am[4][2][16];    // |.|-term W_max
  __shared__ float an[4][2][16];    // |.|-term W_min
#ifdef ASYNC_BMULT
  __shared__ alignas(32) __bf16 BmStage[2][2048];  // 2 x 4KB (4 tiles per d)
#endif
  int w = threadIdx.x >> 5, lane = threadIdx.x & 31;
  int tile = blockIdx.x * 4 + w;
  int h = lane >> 4, n = lane & 15;

  for (int z = lane; z < 32; z += 32) {
    qm[w][z >> 4][z & 15] = 0.f;
    am[w][z >> 4][z & 15] = 0.f;
    an[w][z >> 4][z & 15] = 0.f;
  }
  __syncthreads();

  float mw0 = mixw[0], mw1 = mixw[1], mw2 = mixw[2], mw3 = mixw[3], mw4 = mixw[4];

  // ---- all linear terms as one GEMM: (16 x 1408) * (1408 x 8) ----
  v8f acc = {};
  const v16bf* Ap = (const v16bf*)(Apack + (size_t)tile * NCH * 512);
  const v16bf* Bp = (const v16bf*)Bpair;
  for (int c = 0; c < NCH; ++c) {
    v16bf a  = Ap[c * 32 + lane];
    v16bf bb = Bp[c * 32 + lane];
    acc = __builtin_amdgcn_wmma_f32_16x16x32_bf16(false, a, false, bb,
                                                  (short)0, acc, false, false);
  }
#pragma unroll
  for (int r = 0; r < 8; ++r) Dst[w][r + 8 * h][n] = acc[r];

  // ---- S_mult: per d, U = T_tile * W_mult^T via WMMA, then diag dot ----
  if (mw1 != 0.f) {
    float q0[8], q1[8];
#pragma unroll
    for (int r = 0; r < 8; ++r) { q0[r] = 0.f; q1[r] = 0.f; }
    const v16bf* Am = (const v16bf*)(Amult + (size_t)tile * DIM * 512);
    const v16bf* Bm = (const v16bf*)Bmult;
    (void)Bm;
#ifdef ASYNC_BMULT
    // Block-wide async double-buffered staging of the 4KB/d B-operand tiles.
    auto issue = [&](int dd, int bufi) {
      const char* gsrc = (const char*)Bmult + (size_t)dd * 4096;
      char* lbase = (char*)&BmStage[bufi][0];
#pragma unroll
      for (int s = 0; s < 2; ++s) {
        int c16 = (int)threadIdx.x + 128 * s;
        __builtin_amdgcn_global_load_async_to_lds_b128(
            (gv4i_p)(unsigned long long)(gsrc + c16 * 16),
            (lv4i_p)(unsigned int)(unsigned long long)(lbase + c16 * 16),
            0, 0);
      }
    };
    issue(0, 0);
#endif
    for (int d = 0; d < DIM; ++d) {
#ifdef ASYNC_BMULT
#if __has_builtin(__builtin_amdgcn_s_wait_asynccnt)
      __builtin_amdgcn_s_wait_asynccnt(0);
#else
      asm volatile("s_wait_asynccnt 0x0" ::: "memory");
#endif
      __syncthreads();                       // buf[d&1] ready for all waves
      if (d + 1 < DIM) issue(d + 1, (d + 1) & 1);
      const v16bf* Bl = (const v16bf*)&BmStage[d & 1][0];
#else
      const v16bf* Bl = Bm + d * 4 * 32;
#endif
      v16bf a = Am[d * 32 + lane];
#pragma unroll
      for (int it = 0; it < 2; ++it) {
        v8f tv = *(const v8f*)(TcD +
                 ((((size_t)tile * DIM + d) * 2 + it) * 32 + lane) * 8);
        v8f z0 = {};
        v8f D0 = __builtin_amdgcn_wmma_f32_16x16x32_bf16(
            false, a, false, Bl[(0 * 2 + it) * 32 + lane], (short)0, z0,
            false, false);
        v8f z1 = {};
        v8f D1 = __builtin_amdgcn_wmma_f32_16x16x32_bf16(
            false, a, false, Bl[(1 * 2 + it) * 32 + lane], (short)0, z1,
            false, false);
#pragma unroll
        for (int r = 0; r < 8; ++r) {
          q0[r] += D0[r] * tv[r];
          q1[r] += D1[r] * tv[r];
        }
      }
    }
    // reduce over N (i) across the 16 lanes of each half
#pragma unroll
    for (int off = 1; off < 16; off <<= 1) {
#pragma unroll
      for (int r = 0; r < 8; ++r) {
        q0[r] += __shfl_xor(q0[r], off, 32);
        q1[r] += __shfl_xor(q1[r], off, 32);
      }
    }
    if (n == 0) {
#pragma unroll
      for (int r = 0; r < 8; ++r) {
        qm[w][0][r + 8 * h] = q0[r];
        qm[w][1][r + 8 * h] = q1[r];
      }
    }
  }

  // ---- |T_i - T_j| quadratic terms (VALU; skipped when mix weights are 0) --
  if (mw2 != 0.f || mw3 != 0.f) {
    for (int bb = 0; bb < 16; ++bb) {
      int b = tile * 16 + bb;
      float sx0 = 0, sx1 = 0, sn0 = 0, sn1 = 0;
      for (int p = lane; p < COLS * COLS; p += 32) {
        int i = p / COLS, j = p % COLS;
        const float* Ti = Tf32 + (size_t)b * KID + i * DIM;
        const float* Tj = Tf32 + (size_t)b * KID + j * DIM;
        int base = (i * COLS + j) * 2;
        for (int d = 0; d < DIM; ++d) {
          float av = fabsf(Ti[d] - Tj[d]);
          sx0 += av * Wmax[(base + 0) * DIM + d];
          sx1 += av * Wmax[(base + 1) * DIM + d];
          sn0 += av * Wmin[(base + 0) * DIM + d];
          sn1 += av * Wmin[(base + 1) * DIM + d];
        }
      }
#pragma unroll
      for (int off = 1; off < 32; off <<= 1) {
        sx0 += __shfl_xor(sx0, off, 32);
        sx1 += __shfl_xor(sx1, off, 32);
        sn0 += __shfl_xor(sn0, off, 32);
        sn1 += __shfl_xor(sn1, off, 32);
      }
      if (lane == 0) {
        am[w][0][bb] = sx0; am[w][1][bb] = sx1;
        an[w][0][bb] = sn0; an[w][1][bb] = sn1;
      }
    }
  }
  __syncthreads();

  // ---- combine + loss partial ----
  float wl = 0.f;
  if (lane < 16) {
    int m = lane, b = tile * 16 + m;
#pragma unroll
    for (int k = 0; k < 2; ++k) {
      float inf = mw0 * Dst[w][m][k]
                + mw1 * qm[w][k][m]
                + mw2 * 0.5f * (Dst[w][m][4 + k] + am[w][k][m])
                + mw3 * 0.5f * (Dst[w][m][6 + k] - an[w][k][m])
                + mw4 * Dst[w][m][2 + k];
      out[b * 2 + k] = inf;
      float dlt = inf - label[b * 2 + k];
      wl += posw[b * 2 + k] * dlt * dlt;
    }
  }
#pragma unroll
  for (int off = 1; off < 32; off <<= 1) wl += __shfl_xor(wl, off, 32);
  if (lane == 0) partials[tile] = wl;
}

// Fixed-order scalar reduction -> deterministic weighted_loss.
__global__ void k_loss(const float* __restrict__ partials, float* __restrict__ out) {
  if (threadIdx.x == 0 && blockIdx.x == 0) {
    float s = 0.f;
    for (int t = 0; t < NTILE; ++t) s += partials[t];
    out[BATCH * 2] = s / (float)BATCH;
  }
}

extern "C" void kernel_launch(void* const* d_in, const int* in_sizes, int n_in,
                              void* d_out, int out_size, void* d_ws, size_t ws_size,
                              hipStream_t stream) {
  const int*   fidx  = (const int*)d_in[0];
  const float* label = (const float*)d_in[1];
  const float* posw  = (const float*)d_in[2];
  const float* emb   = (const float*)d_in[3];
  const float* w1    = (const float*)d_in[4];
  const float* b1    = (const float*)d_in[5];
  const float* w2    = (const float*)d_in[6];
  const float* b2    = (const float*)d_in[7];
  const float* Wp    = (const float*)d_in[8];
  const float* Wm    = (const float*)d_in[9];
  const float* Wx    = (const float*)d_in[10];
  const float* Wn    = (const float*)d_in[11];
  const float* Wc    = (const float*)d_in[12];
  const float* mixw  = (const float*)d_in[13];

  char* ws = (char*)d_ws;
  float*  Tf32     = (float*)(ws + OFF_T);
  __bf16* Apack    = (__bf16*)(ws + OFF_APACK);
  __bf16* Amult    = (__bf16*)(ws + OFF_AMULT);
  __bf16* BpairW   = (__bf16*)(ws + OFF_BPAIR);
  __bf16* BmultW   = (__bf16*)(ws + OFF_BMULT);
  float*  TcD      = (float*)(ws + OFF_TCD);
  float*  partials = (float*)(ws + OFF_PART);
  float*  out      = (float*)d_out;

  k_pack_pair<<<(NCH * 512 + 255) / 256, 256, 0, stream>>>(Wp, Wx, Wn, Wc, BpairW);
  k_pack_mult<<<(DIM * 2 * 2 * 512 + 255) / 256, 256, 0, stream>>>(Wm, BmultW);
  k_T<<<(BATCH * KID + 255) / 256, 256, 0, stream>>>(fidx, emb, w1, b1, w2, b2,
                                                     Tf32, Apack, Amult, TcD);
  k_main<<<NTILE / 4, 128, 0, stream>>>(Tf32, Apack, BpairW, Amult, BmultW, TcD,
                                        Wx, Wn, label, posw, mixw, out, partials);
  k_loss<<<1, 32, 0, stream>>>(partials, out);
}
